// NPAutoEncoder_21139829031563
// MI455X (gfx1250) — matrix operand order repaired and checked by reading, not compiled, not run on hardware
//
#include <hip/hip_runtime.h>
#include <hip/hip_bf16.h>

// ---------------------------------------------------------------------------
// CDNA5 / gfx1250 VQ-VAE decoder pipeline, tensorized end-to-end:
//   1) gather_encoder : z_e_x = encoder_w[x]  (+ bf16 position-major copy)
//   2) prep_codebook / prep_weights : bf16 operands, ||e||^2, N-major weights
//   3) vq_argmin      : bf16 WMMA GEMM [16x512x64] per wave + lane argmin
//   4) zq_gather      : z_q_x (f32 ch-major out) + zqt (bf16 pos-major)
//   5) deconv1 via WMMA GEMM [P,64]x[64,512] + deterministic tap-gather + BN
//   6) deconv2 via WMMA GEMM [P,32]x[32,256] + deterministic tap-gather + BN
//   7) 3x3 conv + sigmoid -> x_tilde
// LDS staging uses GLOBAL_LOAD_ASYNC_TO_LDS_B128 (ASYNCcnt) per ISA ch.10.
// ---------------------------------------------------------------------------

typedef __attribute__((ext_vector_type(8)))  float  v8f;
typedef __attribute__((ext_vector_type(16))) __bf16 v16bf;

union FragBF {            // 16 bf16 lane-elements == 8 VGPRs, loaded as 2x uint4
    uint4  q[2];
    v16bf  v;
};

static __device__ __forceinline__ unsigned short f2bf(float f) {
    unsigned int u = __float_as_uint(f);
    unsigned int r = u + 0x7FFFu + ((u >> 16) & 1u);   // round-to-nearest-even
    return (unsigned short)(r >> 16);
}

// ---- async global->LDS copy (16B per lane), tracked by ASYNCcnt -----------
// GV mode: LDS address comes from the vdst VGPR; global addr = 64-bit VGPR pair.
// LDS byte offset = low 32 bits of the generic (flat) shared-pointer value.
static __device__ __forceinline__ void async_b128(void* lds, const void* g) {
    unsigned l = (unsigned)(size_t)lds;
    asm volatile("global_load_async_to_lds_b128 %0, %1, off"
                 :: "v"(l), "v"(g) : "memory");
}
static __device__ __forceinline__ void async_wait0() {
    asm volatile("s_wait_asynccnt 0x0" ::: "memory");
}

// A-fragment (16xK tile row): chunks at +kb and +kb+16 (ISA 7.12.2 16-bit A)
static __device__ __forceinline__ FragBF load_fragA(const unsigned short* row, int kb) {
    FragBF f;
    f.q[0] = *(const uint4*)(row + kb);
    f.q[1] = *(const uint4*)(row + kb + 16);
    return f;
}
// B-fragment (Kx16 tile column, K-contiguous row storage): chunks at +cb and +cb+16
static __device__ __forceinline__ FragBF load_fragB(const unsigned short* row, int cb) {
    FragBF f;
    f.q[0] = *(const uint4*)(row + cb);
    f.q[1] = *(const uint4*)(row + cb + 16);
    return f;
}

// ---------------------------------------------------------------------------
// 1) Embedding gather: z_e_x (f32, [B,64,7,7]) and z_t (bf16, [P,64] pos-major)
// ---------------------------------------------------------------------------
__global__ __launch_bounds__(256) void gather_encoder_kernel(
    const int* __restrict__ x, const float* __restrict__ enc,
    float* __restrict__ ze, unsigned short* __restrict__ zt)
{
    int b = blockIdx.x, tid = threadIdx.x;
    const float* row = enc + (long)x[b] * 3136;
    if (tid == 0) __builtin_prefetch(row, 0, 3);       // global_prefetch_b8
    for (int i = tid; i < 3136; i += 256) {
        float v = row[i];
        ze[(long)b * 3136 + i] = v;
        int c = i / 49, s = i - c * 49;
        zt[((long)b * 49 + s) * 64 + c] = f2bf(v);
    }
}

// ---------------------------------------------------------------------------
// 2a) Codebook prep: bf16 copy + squared norms (512 threads total)
// ---------------------------------------------------------------------------
__global__ __launch_bounds__(256) void prep_codebook_kernel(
    const float* __restrict__ emb, unsigned short* __restrict__ ebf,
    float* __restrict__ esq)
{
    int g = blockIdx.x * 256 + threadIdx.x;            // 0..511
    for (int i = g; i < 512 * 64; i += 512) ebf[i] = f2bf(emb[i]);
    float s = 0.f;
    #pragma unroll
    for (int c = 0; c < 64; ++c) { float v = emb[g * 64 + c]; s += v * v; }
    esq[g] = s;
}

// 2b) Deconv weights -> bf16, N-major rows Wt[n][k] (same layout as codebook).
//     w1 flat = ic*512 + n (n = oc*16+ky*4+kx) ; w2 flat = ic*256 + n.
__global__ __launch_bounds__(256) void prep_weights_kernel(
    const float* __restrict__ w1, const float* __restrict__ w2,
    unsigned short* __restrict__ Wt1, unsigned short* __restrict__ Wt2)
{
    int g = blockIdx.x * 256 + threadIdx.x;            // 0..511
    #pragma unroll
    for (int ic = 0; ic < 64; ++ic)
        Wt1[g * 64 + ic] = f2bf(w1[ic * 512 + g]);
    if (g < 256) {
        #pragma unroll
        for (int ic = 0; ic < 32; ++ic)
            Wt2[g * 32 + ic] = f2bf(w2[ic * 256 + g]);
    }
}

// ---------------------------------------------------------------------------
// 3) VQ argmin via bf16 WMMA.  Block = 8 waves, each wave owns 16 positions.
//    d(n) = ||e_n||^2 - 2 * z.e_n  (||z||^2 irrelevant for argmin)
// ---------------------------------------------------------------------------
__global__ __launch_bounds__(256) void vq_argmin_kernel(
    const unsigned short* __restrict__ zt, const unsigned short* __restrict__ ebf,
    const float* __restrict__ esq, int* __restrict__ latents)
{
    __shared__ __align__(16) unsigned short sE[512 * 64];   // 64 KB codebook
    int tid = threadIdx.x;
    for (int i = tid; i < 4096; i += 256)                   // async 16B/lane
        async_b128(sE + i * 8, ebf + i * 8);
    async_wait0();
    __syncthreads();

    int wave = tid >> 5, lane = tid & 31;
    int half = lane >> 4, l16 = lane & 15;
    long p0 = (long)blockIdx.x * 128;

    const unsigned short* zr = zt + (p0 + wave * 16 + l16) * 64;
    int kb = half * 8;
    FragBF a0 = load_fragA(zr, kb);
    FragBF a1 = load_fragA(zr + 32, kb);

    float best[8]; int bidx[8];
    #pragma unroll
    for (int r = 0; r < 8; ++r) { best[r] = 3.4e38f; bidx[r] = 0; }

    int cb = half * 8;
    #pragma unroll 2
    for (int nt = 0; nt < 32; ++nt) {
        int n = nt * 16 + l16;
        const unsigned short* er = sE + n * 64;
        FragBF b0 = load_fragB(er, cb);
        FragBF b1 = load_fragB(er + 32, cb);

        v8f acc = {};
        acc = __builtin_amdgcn_wmma_f32_16x16x32_bf16(false, a0.v, false, b0.v,
                                                      (short)0, acc, false, false);
        acc = __builtin_amdgcn_wmma_f32_16x16x32_bf16(false, a1.v, false, b1.v,
                                                      (short)0, acc, false, false);
        float eq = esq[n];
        #pragma unroll
        for (int r = 0; r < 8; ++r) {
            float d = eq - 2.0f * acc[r];
            if (d < best[r]) { best[r] = d; bidx[r] = n; }
        }
    }

    #pragma unroll
    for (int r = 0; r < 8; ++r) {
        float v = best[r]; int i = bidx[r];
        #pragma unroll
        for (int off = 8; off >= 1; off >>= 1) {
            float ov = __shfl_xor(v, off, 16);
            int   oi = __shfl_xor(i, off, 16);
            if (ov < v || (ov == v && oi < i)) { v = ov; i = oi; }
        }
        if (l16 == 0) latents[p0 + wave * 16 + half * 8 + r] = i;
    }
}

// ---------------------------------------------------------------------------
// 4) z_q_x = emb_w[latents]: f32 channel-major + bf16 position-major
// ---------------------------------------------------------------------------
__global__ __launch_bounds__(256) void zq_gather_kernel(
    const int* __restrict__ lat, const float* __restrict__ emb,
    float* __restrict__ zq, unsigned short* __restrict__ zqt, long total)
{
    long idx = (long)blockIdx.x * 256 + threadIdx.x;
    if (idx >= total) return;
    long b = idx / 3136; int rem = (int)(idx - b * 3136);
    int c = rem / 49, s = rem - c * 49;
    int code = lat[b * 49 + s];
    float v = emb[code * 64 + c];
    zq[idx] = v;
    zqt[(b * 49 + s) * 64 + c] = f2bf(v);
}

// ---------------------------------------------------------------------------
// 5) deconv1 (64->32, k4 s2 p1, 7x7->14x14) as WMMA GEMM + tap gather.
//    T[m][n] = sum_ic x[m][ic]*w[ic][n], n = oc*16+ky*4+kx; then
//    y[oc][oy][ox] = bias + sum_{valid taps} T[iy*7+ix][oc*16+ky*4+kx].
//    N split in halves of 256; dynamic LDS (>64KB, WGP has 320KB).
// ---------------------------------------------------------------------------
#define T1P 260   // padded LDS row stride (floats) to break bank alignment
__global__ __launch_bounds__(256) void deconv1_wmma_kernel(
    const unsigned short* __restrict__ zqt,  // [B*49][64] bf16
    const unsigned short* __restrict__ Wt1,  // [512][64]  bf16
    const float* __restrict__ bias,
    float* __restrict__ y, float* __restrict__ part)
{
    extern __shared__ __align__(16) char dsm[];
    float* sT           = (float*)dsm;                                   // [64][260]
    unsigned short* sW  = (unsigned short*)(dsm + 64 * T1P * 4);         // [256][64]
    float* sRed         = (float*)(dsm + 64 * T1P * 4 + 256 * 64 * 2);
    float* sRed2        = sRed + 256;

    int tid = threadIdx.x, b = blockIdx.x;
    int wave = tid >> 5, lane = tid & 31, half = lane >> 4, l16 = lane & 15;
    int kb = half * 8, cb = half * 8;

    // A fragments: wave owns M-tile mt = wave&3 (rows m = mt*16..+15, pad >48)
    int mt = wave & 3;
    int m  = mt * 16 + l16;
    int mc = (m < 49) ? m : 0;
    const unsigned short* zr = zqt + ((long)b * 49 + mc) * 64;
    FragBF a0 = load_fragA(zr, kb);
    FragBF a1 = load_fragA(zr + 32, kb);

    for (int h = 0; h < 2; ++h) {
        // stage this half's 256 weight rows (32 KB) via async-to-LDS
        for (int i = tid; i < 2048; i += 256)
            async_b128(sW + i * 8, Wt1 + (long)h * 256 * 64 + i * 8);
        async_wait0();
        __syncthreads();

        int nt0 = (wave >> 2) * 8;                      // waves 0-3: nt 0-7, 4-7: 8-15
        for (int jj = 0; jj < 8; ++jj) {
            int nl = (nt0 + jj) * 16 + l16;
            const unsigned short* er = sW + nl * 64;
            FragBF b0 = load_fragB(er, cb);
            FragBF b1 = load_fragB(er + 32, cb);
            v8f acc = {};
            acc = __builtin_amdgcn_wmma_f32_16x16x32_bf16(false, a0.v, false, b0.v,
                                                          (short)0, acc, false, false);
            acc = __builtin_amdgcn_wmma_f32_16x16x32_bf16(false, a1.v, false, b1.v,
                                                          (short)0, acc, false, false);
            #pragma unroll
            for (int r = 0; r < 8; ++r)
                sT[(mt * 16 + half * 8 + r) * T1P + nl] = acc[r];
        }
        __syncthreads();

        // deterministic tap gather: 16 threads per output channel
        int ocl = tid >> 4, j = tid & 15;
        float bo = bias[h * 16 + ocl];
        float psum = 0.f, psq = 0.f;
        for (int k2 = 0; k2 < 13; ++k2) {
            int s = j + (k2 << 4);
            if (s >= 196) break;
            int oy = s / 14, ox = s - oy * 14;
            float acc = bo;
            #pragma unroll
            for (int ky = 0; ky < 4; ++ky) {
                int ty = oy + 1 - ky;
                if (ty & 1) continue;
                int iy = ty >> 1;
                if ((unsigned)iy >= 7u) continue;
                #pragma unroll
                for (int kx = 0; kx < 4; ++kx) {
                    int tx = ox + 1 - kx;
                    if (tx & 1) continue;
                    int ix = tx >> 1;
                    if ((unsigned)ix >= 7u) continue;
                    acc += sT[(iy * 7 + ix) * T1P + ocl * 16 + ky * 4 + kx];
                }
            }
            y[(long)b * 6272 + (h * 16 + ocl) * 196 + s] = acc;
            psum += acc; psq += acc * acc;
        }
        sRed[tid] = psum; sRed2[tid] = psq;
        __syncthreads();
        if (j == 0) {
            float a = 0.f, q = 0.f;
            for (int t = 0; t < 16; ++t) { a += sRed[tid + t]; q += sRed2[tid + t]; }
            part[(long)b * 64 + h * 16 + ocl]      = a;
            part[(long)b * 64 + 32 + h * 16 + ocl] = q;
        }
        __syncthreads();
    }
}

// ---------------------------------------------------------------------------
// 6) deconv2 (32->16, k4 s2 p1, 14x14->28x28) as WMMA GEMM (K=32, 1 WMMA/tile)
//    N split in quarters of 64 (4 oc at a time); static LDS (~61 KB).
// ---------------------------------------------------------------------------
#define T2P 68
__global__ __launch_bounds__(256) void deconv2_wmma_kernel(
    const unsigned short* __restrict__ h1t,  // [B*196][32] bf16
    const unsigned short* __restrict__ Wt2,  // [256][32]   bf16
    const float* __restrict__ bias,
    float* __restrict__ y, float* __restrict__ part)
{
    __shared__ __align__(16) float sT[208 * T2P];          // 56.6 KB
    __shared__ __align__(16) unsigned short sW[64 * 32];   // 4 KB
    __shared__ float sRed[256], sRed2[256];
    int tid = threadIdx.x, b = blockIdx.x;
    int wave = tid >> 5, lane = tid & 31, half = lane >> 4, l16 = lane & 15;
    int kb = half * 8, cb = half * 8;

    for (int q = 0; q < 4; ++q) {
        if (tid < 128)
            async_b128(sW + tid * 8, Wt2 + (long)q * 64 * 32 + tid * 8);
        async_wait0();
        __syncthreads();

        for (int job = wave; job < 52; job += 8) {         // 13 Mtiles x 4 Ntiles
            int mt = job >> 2, ntl = job & 3;
            int m  = mt * 16 + l16;
            int mc = (m < 196) ? m : 0;
            const unsigned short* zr = h1t + ((long)b * 196 + mc) * 32;
            FragBF a = load_fragA(zr, kb);
            int nl = ntl * 16 + l16;
            const unsigned short* er = sW + nl * 32;
            FragBF bb = load_fragB(er, cb);
            v8f acc = {};
            acc = __builtin_amdgcn_wmma_f32_16x16x32_bf16(false, a.v, false, bb.v,
                                                          (short)0, acc, false, false);
            #pragma unroll
            for (int r = 0; r < 8; ++r)
                sT[(mt * 16 + half * 8 + r) * T2P + nl] = acc[r];
        }
        __syncthreads();

        int ocl = tid >> 6, j = tid & 63;                  // 64 threads per channel
        float bo = bias[q * 4 + ocl];
        float psum = 0.f, psq = 0.f;
        for (int k2 = 0; k2 < 13; ++k2) {
            int s = j + (k2 << 6);
            if (s >= 784) break;
            int oy = s / 28, ox = s - oy * 28;
            float acc = bo;
            #pragma unroll
            for (int ky = 0; ky < 4; ++ky) {
                int ty = oy + 1 - ky;
                if (ty & 1) continue;
                int iy = ty >> 1;
                if ((unsigned)iy >= 14u) continue;
                #pragma unroll
                for (int kx = 0; kx < 4; ++kx) {
                    int tx = ox + 1 - kx;
                    if (tx & 1) continue;
                    int ix = tx >> 1;
                    if ((unsigned)ix >= 14u) continue;
                    acc += sT[(iy * 14 + ix) * T2P + ocl * 16 + ky * 4 + kx];
                }
            }
            y[(long)b * 12544 + (q * 4 + ocl) * 784 + s] = acc;
            psum += acc; psq += acc * acc;
        }
        sRed[tid] = psum; sRed2[tid] = psq;
        __syncthreads();
        if (j == 0) {
            float a = 0.f, qq = 0.f;
            for (int t = 0; t < 64; ++t) { a += sRed[tid + t]; qq += sRed2[tid + t]; }
            part[(long)b * 32 + q * 4 + ocl]      = a;
            part[(long)b * 32 + 16 + q * 4 + ocl] = qq;
        }
        __syncthreads();
    }
}

// ---------------------------------------------------------------------------
// BN helpers: deterministic fixed-order partial reductions, fold to scale/shift,
// fused ReLU (emits pos-major bf16 copy when requested, else f32 in place).
// ---------------------------------------------------------------------------
__global__ __launch_bounds__(256) void reduce_stats_kernel(
    const float* __restrict__ part, float* __restrict__ out, int nBlocks, int width)
{
    __shared__ float red[256];
    int j = blockIdx.x, tid = threadIdx.x;
    float a = 0.f;
    for (int b = tid; b < nBlocks; b += 256) a += part[(long)b * width + j];
    red[tid] = a; __syncthreads();
    for (int st = 128; st > 0; st >>= 1) {
        if (tid < st) red[tid] += red[tid + st];
        __syncthreads();
    }
    if (tid == 0) out[j] = red[0];
}

__global__ void bn_prep_kernel(const float* __restrict__ ss, const float* __restrict__ g,
                               const float* __restrict__ beta, float* __restrict__ scale,
                               float* __restrict__ shift, int C, float invCnt)
{
    int c = threadIdx.x;
    if (c < C) {
        float mean = ss[c] * invCnt;
        float var  = ss[C + c] * invCnt - mean * mean;
        float sc   = g[c] * rsqrtf(var + 1e-5f);
        scale[c] = sc;
        shift[c] = beta[c] - mean * sc;
    }
}

__global__ __launch_bounds__(256) void bn_relu_kernel(
    float* __restrict__ y, unsigned short* __restrict__ ht,
    const float* __restrict__ scale, const float* __restrict__ shift,
    int C, int HW, long total)
{
    long idx = (long)blockIdx.x * 256 + threadIdx.x;
    if (idx >= total) return;
    long chw = (long)C * HW;
    long b = idx / chw;
    int rem = (int)(idx - b * chw);
    int c = rem / HW, s = rem - c * HW;
    float v = y[idx] * scale[c] + shift[c];
    v = v > 0.f ? v : 0.f;
    if (ht) ht[((long)b * HW + s) * C + c] = f2bf(v);   // bf16 pos-major for WMMA
    else    y[idx] = v;                                  // f32 in place
}

// ---------------------------------------------------------------------------
// 7) 3x3 conv 16->1 (pad 1) + sigmoid
// ---------------------------------------------------------------------------
__global__ __launch_bounds__(256) void final_conv_kernel(
    const float* __restrict__ h2, const float* __restrict__ w,
    const float* __restrict__ bc, float* __restrict__ out)
{
    __shared__ __align__(16) float sIn[16 * 784];   // ~50 KB
    __shared__ float sW[144];
    __shared__ float sB;
    int b = blockIdx.x, tid = threadIdx.x;
    for (int i = tid; i < 3136; i += 256)                 // 16*784/4 = 3136 x 16B
        async_b128(sIn + i * 4, h2 + (long)b * 12544 + i * 4);
    if (tid < 144) sW[tid] = w[tid];
    if (tid == 0) sB = bc[0];
    async_wait0();
    __syncthreads();
    for (int s = tid; s < 784; s += 256) {
        int oy = s / 28, ox = s - oy * 28;
        float acc = sB;
        #pragma unroll
        for (int ic = 0; ic < 16; ++ic)
            for (int ky = 0; ky < 3; ++ky) {
                int iy = oy + ky - 1;
                if ((unsigned)iy >= 28u) continue;
                for (int kx = 0; kx < 3; ++kx) {
                    int ix = ox + kx - 1;
                    if ((unsigned)ix >= 28u) continue;
                    acc = fmaf(sIn[ic * 784 + iy * 28 + ix], sW[ic * 9 + ky * 3 + kx], acc);
                }
            }
        out[(long)b * 784 + s] = 1.f / (1.f + __expf(-acc));
    }
}

// ---------------------------------------------------------------------------
extern "C" void kernel_launch(void* const* d_in, const int* in_sizes, int n_in,
                              void* d_out, int out_size, void* d_ws, size_t ws_size,
                              hipStream_t stream)
{
    const int*   x   = (const int*)  d_in[0];
    const float* enc = (const float*)d_in[1];
    const float* emb = (const float*)d_in[2];
    const float* w1  = (const float*)d_in[3];
    const float* b1  = (const float*)d_in[4];
    const float* g1  = (const float*)d_in[5];
    const float* be1 = (const float*)d_in[6];
    const float* w2  = (const float*)d_in[7];
    const float* b2  = (const float*)d_in[8];
    const float* g2  = (const float*)d_in[9];
    const float* be2 = (const float*)d_in[10];
    const float* wc  = (const float*)d_in[11];
    const float* bc  = (const float*)d_in[12];
    (void)n_in; (void)out_size; (void)ws_size;

    const int  B = in_sizes[0];          // 4096
    const long P = (long)B * 49;         // 200704

    float* out = (float*)d_out;
    float* xt  = out;                    // [B,1,28,28]
    float* ze  = out + (long)B * 784;    // [B,64,7,7]
    float* zq  = ze  + (long)B * 3136;   // [B,64,7,7]

    char* ws = (char*)d_ws;
    size_t off = 0;
    auto carve = [&](size_t bytes) {
        char* p = ws + off;
        off = (off + bytes + 255) & ~(size_t)255;
        return p;
    };
    unsigned short* zt  = (unsigned short*)carve((size_t)P * 64 * 2);     // 25.7 MB
    unsigned short* zqt = (unsigned short*)carve((size_t)P * 64 * 2);     // 25.7 MB
    unsigned short* h1t = (unsigned short*)carve((size_t)B * 196 * 32 * 2); // 51.4 MB
    unsigned short* ebf = (unsigned short*)carve(512 * 64 * 2);
    unsigned short* Wt1 = (unsigned short*)carve(512 * 64 * 2);
    unsigned short* Wt2 = (unsigned short*)carve(256 * 32 * 2);
    float* esq   = (float*)carve(512 * 4);
    int*   lat   = (int*)  carve((size_t)P * 4);
    float* y1    = (float*)carve((size_t)B * 6272  * 4);                  // 103 MB
    float* y2    = (float*)carve((size_t)B * 12544 * 4);                  // 205 MB
    float* part1 = (float*)carve((size_t)B * 64 * 4);
    float* part2 = (float*)carve((size_t)B * 32 * 4);
    float* st1   = (float*)carve(64 * 4);
    float* sc1   = (float*)carve(32 * 4);
    float* sh1   = (float*)carve(32 * 4);
    float* st2   = (float*)carve(32 * 4);
    float* sc2   = (float*)carve(16 * 4);
    float* sh2   = (float*)carve(16 * 4);

    gather_encoder_kernel<<<B, 256, 0, stream>>>(x, enc, ze, zt);
    prep_codebook_kernel<<<2, 256, 0, stream>>>(emb, ebf, esq);
    prep_weights_kernel<<<2, 256, 0, stream>>>(w1, w2, Wt1, Wt2);
    vq_argmin_kernel<<<(int)(P / 128), 256, 0, stream>>>(zt, ebf, esq, lat);
    {
        long tot = (long)B * 3136;
        zq_gather_kernel<<<(int)((tot + 255) / 256), 256, 0, stream>>>(lat, emb, zq, zqt, tot);
    }
    {
        size_t sh = (size_t)64 * T1P * 4 + 256 * 64 * 2 + 2 * 256 * 4;    // 101,376 B
        deconv1_wmma_kernel<<<B, 256, sh, stream>>>(zqt, Wt1, b1, y1, part1);
    }
    reduce_stats_kernel<<<64, 256, 0, stream>>>(part1, st1, B, 64);
    bn_prep_kernel<<<1, 32, 0, stream>>>(st1, g1, be1, sc1, sh1, 32, 1.0f / ((float)B * 196.f));
    {
        long tot = (long)B * 6272;
        bn_relu_kernel<<<(int)((tot + 255) / 256), 256, 0, stream>>>(y1, h1t, sc1, sh1, 32, 196, tot);
    }
    deconv2_wmma_kernel<<<B, 256, 0, stream>>>(h1t, Wt2, b2, y2, part2);
    reduce_stats_kernel<<<32, 256, 0, stream>>>(part2, st2, B, 32);
    bn_prep_kernel<<<1, 32, 0, stream>>>(st2, g2, be2, sc2, sh2, 16, 1.0f / ((float)B * 784.f));
    {
        long tot = (long)B * 12544;
        bn_relu_kernel<<<(int)((tot + 255) / 256), 256, 0, stream>>>(y2, (unsigned short*)nullptr,
                                                                     sc2, sh2, 16, 784, tot);
    }
    final_conv_kernel<<<B, 256, 0, stream>>>(y2, wc, bc, xt);
}
